// SampledPropagator_12189117186388
// MI455X (gfx1250) — matrix operand order repaired and checked by reading, not compiled, not run on hardware
//
#include <hip/hip_runtime.h>
#include <hip/hip_bf16.h>

#define NN      200000
#define HDIM    64
#define DDEG    64
#define RREL    24
#define MTOP    512
#define KTOP    32
#define NITEMS  50000
#define NENT    150000
#define UOFF    150000
#define NEGV    (-1e9f)
#define NBLK    782                 // ceil(NN/256)
#define NMSG    (MTOP*(KTOP+1))     // 16896 messages (K edges + self loop per expander)

typedef __bf16 v16bf __attribute__((ext_vector_type(16)));
typedef __bf16 v8bf  __attribute__((ext_vector_type(8)));
typedef float  v8f   __attribute__((ext_vector_type(8)));

// ---------------- WMMA helpers (wave32, 16x16x32 bf16 -> f32) ----------------

__device__ inline v8f wmma_bf16(v16bf a, v16bf b, v8f c) {
  return __builtin_amdgcn_wmma_f32_16x16x32_bf16(false, a, false, b, (short)0, c,
                                                 false, false);
}

__device__ inline v16bf cat8(v8bf lo, v8bf hi) {
  return __builtin_shufflevector(lo, hi, 0, 1, 2, 3, 4, 5, 6, 7,
                                 8, 9, 10, 11, 12, 13, 14, 15);
}

// A tiles (16 rows x K=64) straight from a row-major bf16 matrix (ld = 64).
// Lane l: row = l&15, kh = (l&16)?8:0. 16-bit A layout: a0 covers K kh..kh+7 and
// 16+kh..16+kh+7 (two aligned 16B segments); a1 the same at +32. Pure b128 loads.
__device__ inline void ld_aB(const __bf16* __restrict__ hb, int row0,
                             v16bf& a0, v16bf& a1) {
  int lane = threadIdx.x & 31;
  const __bf16* p = hb + (size_t)(row0 + (lane & 15)) * HDIM + ((lane & 16) ? 8 : 0);
  const v8bf* q = (const v8bf*)p;      // 16B aligned: row stride 128B, kh*2 in {0,16}
  a0 = cat8(q[0], q[2]);
  a1 = cat8(q[4], q[6]);
}

// B tile from pre-packed per-lane layout: 32 contiguous bytes per lane per tile.
__device__ inline v16bf ld_bP(const __bf16* __restrict__ packed, int tile) {
  const v8bf* q = (const v8bf*)(packed + ((size_t)tile * 32 + (threadIdx.x & 31)) * 16);
  return cat8(q[0], q[1]);
}

__device__ inline unsigned sortKey(float x) {
  unsigned u = __float_as_uint(x);
  return (u & 0x80000000u) ? ~u : (u | 0x80000000u);
}

__device__ inline float sigm(float x) { return 1.f / (1.f + expf(-x)); }

// ---------------- init / weight packing ----------------

__global__ void init_kernel(float* h, __bf16* hb, int* F) {
  int i = blockIdx.x * 256 + threadIdx.x;
  if (i < NN * HDIM) { h[i] = 0.f; hb[i] = (__bf16)0.f; }
  if (i < NN) F[i] = 0;
}

__global__ void root_kernel(int* F, const int* user_id) {
  F[UOFF + *user_id] = 1;
}

// Pack weight W (row-major [nrows][64], used as W^T in GEMM) into the exact
// per-lane WMMA B-tile layout: tile t=(nt,kc); elem = W[(nt*16+col)*64 + kc*32 + kb + i + (i&8)]
__global__ void pack_b_kernel(const float* __restrict__ W, int ntiles,
                              __bf16* __restrict__ out) {
  int idx = blockIdx.x * 256 + threadIdx.x;
  if (idx >= ntiles * 2 * 512) return;
  int t = idx >> 9;
  int r = idx & 511;
  int lane = r >> 4, i = r & 15;
  int nt = t >> 1, kc = t & 1;
  int col = lane & 15, kb = (lane & 16) ? 8 : 0;
  out[idx] = (__bf16)W[(size_t)(nt * 16 + col) * HDIM + kc * 32 + kb + i + (i & 8)];
}

// rel_gi[r][c] = rel_emb[r] . W_ih[c][0:64];  type_gi[t][c] = type_emb[t] . W_ih[c][64:128] + b_ih[c]
__global__ void tables_kernel(const float* __restrict__ type_emb, const float* __restrict__ rel_emb,
                              const float* __restrict__ W_ih, const float* __restrict__ b_ih,
                              float* rel_gi, float* type_gi) {
  int tid = blockIdx.x * blockDim.x + threadIdx.x;
  int stride = gridDim.x * blockDim.x;
  for (int j = tid; j < RREL * 192; j += stride) {
    int r = j / 192, c = j % 192;
    float s = 0.f;
#pragma unroll
    for (int k = 0; k < HDIM; ++k) s += rel_emb[r * HDIM + k] * W_ih[c * 128 + k];
    rel_gi[j] = s;
  }
  for (int j = tid; j < 4 * 192; j += stride) {
    int t = j / 192, c = j % 192;
    float s = b_ih[c];
#pragma unroll
    for (int k = 0; k < HDIM; ++k) s += type_emb[t * HDIM + k] * W_ih[c * 128 + 64 + k];
    type_gi[j] = s;
  }
}

// ---------------- alpha MLP + masked sort keys (bf16 WMMA, bf16 h shadow) ----------------

__global__ void alpha_keys_kernel(const __bf16* __restrict__ hb,
                                  const __bf16* __restrict__ we1p,
                                  const float* __restrict__ be1,
                                  const float* __restrict__ We2, const float* __restrict__ be2,
                                  const int* __restrict__ F, unsigned* __restrict__ keys) {
  int wave = blockIdx.x * 8 + (threadIdx.x >> 5);
  int row0 = wave * 16;
  if (row0 >= NN) return;
  if (row0 + 16 < NN) __builtin_prefetch(hb + (size_t)(row0 + 16) * HDIM, 0, 0);
  int lane = threadIdx.x & 31;
  int col  = lane & 15;
  v16bf a0, a1;
  ld_aB(hb, row0, a0, a1);
  float s[8] = {};
#pragma unroll
  for (int nt = 0; nt < 4; ++nt) {
    v8f acc = {};
    acc = wmma_bf16(a0, ld_bP(we1p, nt * 2), acc);
    acc = wmma_bf16(a1, ld_bP(we1p, nt * 2 + 1), acc);
    int n = nt * 16 + col;
    float b1v = be1[n], w2 = We2[n];
#pragma unroll
    for (int r = 0; r < 8; ++r) {
      float t = acc[r] + b1v;
      t = t > 0.f ? t : 0.f;
      s[r] += t * w2;
    }
  }
#pragma unroll
  for (int m = 1; m < 16; m <<= 1)
#pragma unroll
    for (int r = 0; r < 8; ++r) s[r] += __shfl_xor(s[r], m, 32);
  if (col == 0) {
    int rb = (lane & 16) ? 8 : 0;
    float b2 = be2[0];
#pragma unroll
    for (int r = 0; r < 8; ++r) {
      int row = row0 + rb + r;
      float v = s[r] + b2;
      keys[row] = sortKey(F[row] ? v : NEGV);
    }
  }
}

// ---------------- radix-select top-M threshold ----------------

__global__ void zero_hist_kernel(unsigned* hist) {
  int i = blockIdx.x * 256 + threadIdx.x;
  if (i < 65536) hist[i] = 0;
}

__global__ void hist_hi_kernel(const unsigned* __restrict__ keys, unsigned* hist) {
  int n = blockIdx.x * 256 + threadIdx.x;
  if (n < NN) atomicAdd(&hist[keys[n] >> 16], 1u);
}

__global__ void pick_hi_kernel(const unsigned* __restrict__ hist, int* meta) {
  __shared__ unsigned cs[256];
  int t = threadIdx.x;
  unsigned s = 0;
  for (int i = 0; i < 256; ++i) s += hist[t * 256 + i];
  cs[t] = s;
  __syncthreads();
  if (t == 0) {
    unsigned cum = 0;
    int c;
    for (c = 255; c >= 0; --c) { if (cum + cs[c] >= MTOP) break; cum += cs[c]; }
    int b;
    for (b = 255; b >= 0; --b) { unsigned hh = hist[c * 256 + b]; if (cum + hh >= MTOP) break; cum += hh; }
    meta[0] = c * 256 + b;   // hi 16 bits of threshold
    meta[1] = (int)cum;      // count strictly above hi bin
  }
}

__global__ void hist_lo_kernel(const unsigned* __restrict__ keys, unsigned* hist,
                               const int* __restrict__ meta) {
  int n = blockIdx.x * 256 + threadIdx.x;
  if (n < NN) {
    unsigned k = keys[n];
    if ((int)(k >> 16) == meta[0]) atomicAdd(&hist[k & 0xFFFFu], 1u);
  }
}

__global__ void pick_lo_kernel(const unsigned* __restrict__ hist, int* meta) {
  __shared__ unsigned cs[256];
  int t = threadIdx.x;
  unsigned s = 0;
  for (int i = 0; i < 256; ++i) s += hist[t * 256 + i];
  cs[t] = s;
  __syncthreads();
  if (t == 0) {
    unsigned cum = (unsigned)meta[1];
    int c;
    for (c = 255; c >= 0; --c) { if (cum + cs[c] >= MTOP) break; cum += cs[c]; }
    int b;
    for (b = 255; b >= 0; --b) { unsigned hh = hist[c * 256 + b]; if (cum + hh >= MTOP) break; cum += hh; }
    meta[2] = (int)(((unsigned)meta[0] << 16) | (unsigned)(c * 256 + b));  // threshold key
  }
}

__global__ void count_kernel(const unsigned* __restrict__ keys, const int* __restrict__ meta,
                             int* blkCnt) {
  __shared__ int cg, ce;
  if (threadIdx.x == 0) { cg = 0; ce = 0; }
  __syncthreads();
  int n = blockIdx.x * 256 + threadIdx.x;
  unsigned thr = (unsigned)meta[2];
  if (n < NN) {
    unsigned k = keys[n];
    if (k > thr) atomicAdd(&cg, 1);
    else if (k == thr) atomicAdd(&ce, 1);
  }
  __syncthreads();
  if (threadIdx.x == 0) { blkCnt[2 * blockIdx.x] = cg; blkCnt[2 * blockIdx.x + 1] = ce; }
}

__global__ void scan_kernel(const int* __restrict__ blkCnt, int* gtBase, int* eqBase, int* meta) {
  int g = 0, e = 0;
  for (int b = 0; b < NBLK; ++b) {
    gtBase[b] = g; g += blkCnt[2 * b];
    eqBase[b] = e; e += blkCnt[2 * b + 1];
  }
  meta[4] = g;          // totalGT
  meta[3] = MTOP - g;   // number of ==thr elements to take (index order)
}

__global__ void scatter_kernel(const unsigned* __restrict__ keys, const int* __restrict__ meta,
                               const int* __restrict__ gtBase, const int* __restrict__ eqBase,
                               int* exp_ids) {
  __shared__ int sg[256], se[256];
  int tid = threadIdx.x;
  int n = blockIdx.x * 256 + tid;
  unsigned thr = (unsigned)meta[2];
  int gt = 0, eq = 0;
  if (n < NN) {
    unsigned k = keys[n];
    gt = (k > thr) ? 1 : 0;
    eq = (k == thr) ? 1 : 0;
  }
  sg[tid] = gt; se[tid] = eq;
  __syncthreads();
  for (int off = 1; off < 256; off <<= 1) {
    int a = (tid >= off) ? sg[tid - off] : 0;
    int b = (tid >= off) ? se[tid - off] : 0;
    __syncthreads();
    sg[tid] += a; se[tid] += b;
    __syncthreads();
  }
  int totalGT = meta[4], takeEq = meta[3];
  if (gt) exp_ids[gtBase[blockIdx.x] + sg[tid] - 1] = n;
  if (eq) {
    int gp = eqBase[blockIdx.x] + se[tid] - 1;
    if (gp < takeEq) exp_ids[totalGT + gp] = n;
  }
}

// ---------------- gather expander state (f32 + bf16 shadow) ----------------

__global__ void gather_kernel(const float* __restrict__ h, const int* __restrict__ F,
                              const int* __restrict__ exp_ids, const int* __restrict__ user_id,
                              int* validv, int* typv, float* h_s, __bf16* h_sb) {
  int idx = blockIdx.x * 256 + threadIdx.x;  // MTOP*HDIM
  int m = idx >> 6, i = idx & 63;
  int eid = exp_ids[m];
  float v = h[(size_t)eid * HDIM + i];
  h_s[idx] = v;
  h_sb[idx] = (__bf16)v;
  if (i == 0) {
    validv[m] = F[eid];
    int root = UOFF + *user_id;
    typv[m] = eid < NITEMS ? 0 : (eid < NENT ? 1 : (eid == root ? 2 : 3));
  }
}

// ---------------- gh = h_s @ W_hh^T + b_hh  (WMMA, [512,64]x[64,192]) ----------------

__global__ void gh_gemm_kernel(const __bf16* __restrict__ h_sb, const __bf16* __restrict__ whhp,
                               const float* __restrict__ b_hh, float* __restrict__ ghb) {
  int wave = blockIdx.x * 8 + (threadIdx.x >> 5);   // 32 mtiles * 12 ntiles = 384 waves
  if (wave >= 384) return;
  int mt = wave / 12, nt = wave % 12;
  int lane = threadIdx.x & 31, col = lane & 15;
  v16bf a0, a1;
  ld_aB(h_sb, mt * 16, a0, a1);
  v8f acc = {};
  acc = wmma_bf16(a0, ld_bP(whhp, nt * 2), acc);
  acc = wmma_bf16(a1, ld_bP(whhp, nt * 2 + 1), acc);
  int n = nt * 16 + col;
  float bb = b_hh[n];
  int rb = (lane & 16) ? 8 : 0;
#pragma unroll
  for (int r = 0; r < 8; ++r) ghb[(size_t)(mt * 16 + rb + r) * 192 + n] = acc[r] + bb;
}

// ---------------- GRU gates (input GEMM factored into rel/type tables) ----------------

__global__ void gru_kernel(const int* __restrict__ exp_ids, const int* __restrict__ rel_ids,
                           const int* __restrict__ typv, const float* __restrict__ rel_gi,
                           const float* __restrict__ type_gi, const float* __restrict__ ghb,
                           const float* __restrict__ h_s, float* __restrict__ g,
                           __bf16* __restrict__ gb) {
  int idx = blockIdx.x * 256 + threadIdx.x;   // MTOP*DDEG*HDIM
  int m = idx >> 12, d = (idx >> 6) & 63, hd = idx & 63;
  int eid = exp_ids[m];
  int rel = rel_ids[(size_t)eid * DDEG + d];
  const float* rg = rel_gi + rel * 192;
  const float* tg = type_gi + typv[m] * 192;
  const float* gh = ghb + (size_t)m * 192;
  float gi0 = rg[hd]       + tg[hd];
  float gi1 = rg[64 + hd]  + tg[64 + hd];
  float gi2 = rg[128 + hd] + tg[128 + hd];
  float r = sigm(gi0 + gh[hd]);
  float z = sigm(gi1 + gh[64 + hd]);
  float n = tanhf(gi2 + r * gh[128 + hd]);
  float o = (1.f - z) * n + z * h_s[(size_t)m * HDIM + hd];
  g[idx] = o;
  gb[idx] = (__bf16)o;
}

// ---------------- beta MLP (WMMA over [32768,64], bf16 g shadow) ----------------

__global__ void beta_gemm_kernel(const __bf16* __restrict__ gb, const __bf16* __restrict__ wn1p,
                                 const float* __restrict__ bn1, const float* __restrict__ Wn2,
                                 const float* __restrict__ bn2, float* __restrict__ beta) {
  int wave = blockIdx.x * 8 + (threadIdx.x >> 5);  // 2048 waves, 16 rows each
  int row0 = wave * 16;
  int lane = threadIdx.x & 31, col = lane & 15;
  v16bf a0, a1;
  ld_aB(gb, row0, a0, a1);
  float s[8] = {};
#pragma unroll
  for (int nt = 0; nt < 4; ++nt) {
    v8f acc = {};
    acc = wmma_bf16(a0, ld_bP(wn1p, nt * 2), acc);
    acc = wmma_bf16(a1, ld_bP(wn1p, nt * 2 + 1), acc);
    int n = nt * 16 + col;
    float b1v = bn1[n], w2 = Wn2[n];
#pragma unroll
    for (int r = 0; r < 8; ++r) {
      float t = acc[r] + b1v;
      t = t > 0.f ? t : 0.f;
      s[r] += t * w2;
    }
  }
#pragma unroll
  for (int m = 1; m < 16; m <<= 1)
#pragma unroll
    for (int r = 0; r < 8; ++r) s[r] += __shfl_xor(s[r], m, 32);
  if (col == 0) {
    int rb = (lane & 16) ? 8 : 0;
    float b2 = bn2[0];
#pragma unroll
    for (int r = 0; r < 8; ++r) beta[row0 + rb + r] = s[r] + b2;
  }
}

// ---------------- per-expander top-K (rank == output slot, deterministic) ----------------

__global__ void topk_kernel(const float* __restrict__ beta, int* __restrict__ kidx) {
  __shared__ float sb[4 * 64];
  int tid = threadIdx.x;
  int lm = tid >> 6, d = tid & 63;
  int m = blockIdx.x * 4 + lm;
  float v = beta[(size_t)m * DDEG + d];
  sb[lm * 64 + d] = v;
  __syncthreads();
  int rank = 0;
#pragma unroll
  for (int dd = 0; dd < DDEG; ++dd) {
    float o = sb[lm * 64 + dd];
    rank += (o > v) || (o == v && dd < d);
  }
  if (rank < KTOP) kidx[(size_t)m * KTOP + rank] = d;
}

// ---------------- frontier clear (expanded nodes leave F) ----------------

__global__ void fclear_kernel(const int* __restrict__ exp_ids, const int* __restrict__ validv,
                              int* F) {
  int m = blockIdx.x * 256 + threadIdx.x;
  if (m < MTOP && validv[m]) F[exp_ids[m]] = 0;
}

// ---------------- message scatter: init / accumulate exp / finalize log ----------------

__device__ inline bool msg_decode(int e, const int* __restrict__ exp_ids,
                                  const int* __restrict__ validv, const int* __restrict__ kidx,
                                  const int* __restrict__ neighbors,
                                  int& dst, size_t& gofs, bool& self) {
  if (e < MTOP * KTOP) {
    int m = e >> 5, j = e & 31;
    if (!validv[m]) return false;
    int d = kidx[(size_t)m * KTOP + j];
    int eid = exp_ids[m];
    dst = neighbors[(size_t)eid * DDEG + d];
    gofs = (size_t)m * DDEG * HDIM + (size_t)d * HDIM;
    self = false;
    return true;
  }
  int m = e - MTOP * KTOP;
  if (!validv[m]) return false;
  dst = exp_ids[m];
  gofs = (size_t)m * HDIM;
  self = true;
  return true;
}

__global__ void msg_init_kernel(const int* __restrict__ exp_ids, const int* __restrict__ validv,
                                const int* __restrict__ kidx, const int* __restrict__ neighbors,
                                float* __restrict__ sumbuf) {
  int idx = blockIdx.x * 256 + threadIdx.x;   // NMSG*HDIM
  int e = idx >> 6, hd = idx & 63;
  int dst; size_t gofs; bool self;
  if (msg_decode(e, exp_ids, validv, kidx, neighbors, dst, gofs, self))
    sumbuf[(size_t)dst * HDIM + hd] = 0.f;
}

__global__ void msg_accum_kernel(const int* __restrict__ exp_ids, const int* __restrict__ validv,
                                 const int* __restrict__ kidx, const int* __restrict__ neighbors,
                                 const float* __restrict__ g, const float* __restrict__ h_s,
                                 float* __restrict__ sumbuf) {
  int idx = blockIdx.x * 256 + threadIdx.x;
  int e = idx >> 6, hd = idx & 63;
  int dst; size_t gofs; bool self;
  if (msg_decode(e, exp_ids, validv, kidx, neighbors, dst, gofs, self)) {
    float msg = self ? h_s[gofs + hd] : g[gofs + hd];
    atomicAdd(&sumbuf[(size_t)dst * HDIM + hd], expf(msg));   // global_atomic_add_f32
  }
}

__global__ void msg_final_kernel(const int* __restrict__ exp_ids, const int* __restrict__ validv,
                                 const int* __restrict__ kidx, const int* __restrict__ neighbors,
                                 const float* __restrict__ sumbuf, float* __restrict__ h,
                                 __bf16* __restrict__ hb, int* F) {
  int idx = blockIdx.x * 256 + threadIdx.x;
  int e = idx >> 6, hd = idx & 63;
  int dst; size_t gofs; bool self;
  if (msg_decode(e, exp_ids, validv, kidx, neighbors, dst, gofs, self)) {
    float v = sumbuf[(size_t)dst * HDIM + hd];
    float o = logf(fmaxf(v, 1e-30f));                         // logsumexp (mx folded out)
    h[(size_t)dst * HDIM + hd] = o;
    hb[(size_t)dst * HDIM + hd] = (__bf16)o;                  // keep bf16 shadow coherent
    if (hd == 0) F[dst] = 1;                                  // received -> in next frontier
  }
}

// ---------------- launcher ----------------

extern "C" void kernel_launch(void* const* d_in, const int* in_sizes, int n_in,
                              void* d_out, int out_size, void* d_ws, size_t ws_size,
                              hipStream_t stream) {
  const float* type_emb = (const float*)d_in[0];
  const float* rel_emb  = (const float*)d_in[1];
  const float* W_ih     = (const float*)d_in[2];
  const float* W_hh     = (const float*)d_in[3];
  const float* b_ih     = (const float*)d_in[4];
  const float* b_hh     = (const float*)d_in[5];
  const float* We1      = (const float*)d_in[6];
  const float* be1      = (const float*)d_in[7];
  const float* We2      = (const float*)d_in[8];
  const float* be2      = (const float*)d_in[9];
  const float* Wn1      = (const float*)d_in[10];
  const float* bn1      = (const float*)d_in[11];
  const float* Wn2      = (const float*)d_in[12];
  const float* bn2      = (const float*)d_in[13];
  const int* neighbors  = (const int*)d_in[14];
  const int* rel_ids    = (const int*)d_in[15];
  const int* user_id    = (const int*)d_in[16];
  // d_in[17] = num_layers (== 3 in setup; unrolled here for capture determinism)

  float* h = (float*)d_out;   // h state lives in the output buffer

  char* w = (char*)d_ws;
  auto alloc = [&](size_t bytes) -> char* {
    char* p = w;
    w += (bytes + 255) & ~(size_t)255;
    return p;
  };
  unsigned* keys  = (unsigned*)alloc((size_t)NN * 4);
  unsigned* hist  = (unsigned*)alloc(65536 * 4);
  int* meta       = (int*)alloc(256);
  int* blkCnt     = (int*)alloc((size_t)NBLK * 2 * 4);
  int* gtBase     = (int*)alloc((size_t)NBLK * 4);
  int* eqBase     = (int*)alloc((size_t)NBLK * 4);
  int* exp_ids    = (int*)alloc(MTOP * 4);
  int* validv     = (int*)alloc(MTOP * 4);
  int* typv       = (int*)alloc(MTOP * 4);
  float* h_s      = (float*)alloc((size_t)MTOP * HDIM * 4);
  __bf16* h_sb    = (__bf16*)alloc((size_t)MTOP * HDIM * 2);
  float* ghb      = (float*)alloc((size_t)MTOP * 192 * 4);
  float* rel_gi   = (float*)alloc((size_t)RREL * 192 * 4);
  float* type_gi  = (float*)alloc((size_t)4 * 192 * 4);
  float* gbuf     = (float*)alloc((size_t)MTOP * DDEG * HDIM * 4);
  __bf16* gbuf_b  = (__bf16*)alloc((size_t)MTOP * DDEG * HDIM * 2);
  float* betab    = (float*)alloc((size_t)MTOP * DDEG * 4);
  int* kidx       = (int*)alloc((size_t)MTOP * KTOP * 4);
  int* F          = (int*)alloc((size_t)NN * 4);
  __bf16* hbf     = (__bf16*)alloc((size_t)NN * HDIM * 2);   // bf16 shadow of h
  __bf16* we1p    = (__bf16*)alloc((size_t)8 * 512 * 2);     // packed We1   (4 ntiles)
  __bf16* whhp    = (__bf16*)alloc((size_t)24 * 512 * 2);    // packed W_hh  (12 ntiles)
  __bf16* wn1p    = (__bf16*)alloc((size_t)8 * 512 * 2);     // packed Wn1   (4 ntiles)
  float* sumbuf   = (float*)alloc((size_t)NN * HDIM * 4);

  init_kernel<<<(NN * HDIM + 255) / 256, 256, 0, stream>>>(h, hbf, F);
  root_kernel<<<1, 1, 0, stream>>>(F, user_id);
  tables_kernel<<<8, 256, 0, stream>>>(type_emb, rel_emb, W_ih, b_ih, rel_gi, type_gi);
  pack_b_kernel<<<16, 256, 0, stream>>>(We1, 4, we1p);
  pack_b_kernel<<<48, 256, 0, stream>>>(W_hh, 12, whhp);
  pack_b_kernel<<<16, 256, 0, stream>>>(Wn1, 4, wn1p);

  for (int layer = 0; layer < 3; ++layer) {
    alpha_keys_kernel<<<1563, 256, 0, stream>>>(hbf, we1p, be1, We2, be2, F, keys);

    zero_hist_kernel<<<256, 256, 0, stream>>>(hist);
    hist_hi_kernel<<<NBLK, 256, 0, stream>>>(keys, hist);
    pick_hi_kernel<<<1, 256, 0, stream>>>(hist, meta);
    zero_hist_kernel<<<256, 256, 0, stream>>>(hist);
    hist_lo_kernel<<<NBLK, 256, 0, stream>>>(keys, hist, meta);
    pick_lo_kernel<<<1, 256, 0, stream>>>(hist, meta);
    count_kernel<<<NBLK, 256, 0, stream>>>(keys, meta, blkCnt);
    scan_kernel<<<1, 1, 0, stream>>>(blkCnt, gtBase, eqBase, meta);
    scatter_kernel<<<NBLK, 256, 0, stream>>>(keys, meta, gtBase, eqBase, exp_ids);

    gather_kernel<<<(MTOP * HDIM) / 256, 256, 0, stream>>>(h, F, exp_ids, user_id,
                                                           validv, typv, h_s, h_sb);
    gh_gemm_kernel<<<48, 256, 0, stream>>>(h_sb, whhp, b_hh, ghb);
    gru_kernel<<<(MTOP * DDEG * HDIM) / 256, 256, 0, stream>>>(exp_ids, rel_ids, typv, rel_gi,
                                                               type_gi, ghb, h_s, gbuf, gbuf_b);
    beta_gemm_kernel<<<256, 256, 0, stream>>>(gbuf_b, wn1p, bn1, Wn2, bn2, betab);
    topk_kernel<<<MTOP / 4, 256, 0, stream>>>(betab, kidx);

    fclear_kernel<<<2, 256, 0, stream>>>(exp_ids, validv, F);
    msg_init_kernel<<<(NMSG * HDIM) / 256, 256, 0, stream>>>(exp_ids, validv, kidx,
                                                             neighbors, sumbuf);
    msg_accum_kernel<<<(NMSG * HDIM) / 256, 256, 0, stream>>>(exp_ids, validv, kidx, neighbors,
                                                              gbuf, h_s, sumbuf);
    msg_final_kernel<<<(NMSG * HDIM) / 256, 256, 0, stream>>>(exp_ids, validv, kidx, neighbors,
                                                              sumbuf, h, hbf, F);
  }
}
